// HRTNoEdgeInit_35338990911499
// MI455X (gfx1250) — matrix operand order repaired and checked by reading, not compiled, not run on hardware
//
#include <hip/hip_runtime.h>

// ---------------------------------------------------------------------------
// Types / WMMA helper (CDNA5 wave32: V_WMMA_F32_16X16X32_BF16)
// ---------------------------------------------------------------------------
typedef __attribute__((ext_vector_type(16))) __bf16 v16bf;
typedef __attribute__((ext_vector_type(8)))  float  v8f;

__device__ __forceinline__ v8f wmma_bf16(v16bf a, v16bf b, v8f c) {
  // 8 args: (neg_a, A, neg_b, B, c_mod, C, reuse_a, reuse_b)
  return __builtin_amdgcn_wmma_f32_16x16x32_bf16(false, a, false, b, (short)0, c,
                                                 false, false);
}

__device__ __forceinline__ unsigned int pk2bf(float a, float b) {
  unsigned short ua = __builtin_bit_cast(unsigned short, (__bf16)a);
  unsigned short ub = __builtin_bit_cast(unsigned short, (__bf16)b);
  return (unsigned int)ua | ((unsigned int)ub << 16);
}

// A-fragment (16x32 bf16, M x K) element j of a lane:
//   row = lane&15 ; kg = lane>>4 ; k = kg*16 + j
// B-fragment (32x16 bf16, K x N) element j of a lane:
//   col = lane&15 ; kg = lane>>4 ; k = kg*16 + j
// C/D (16x16 f32): row = (lane>>4)*8 + i ; col = lane&15

// ---------------------------------------------------------------------------
// Generic tiled GEMM:  out[b] (+)= act( A[b] @ W[b] + bias )
//   A: [M,K] f32 (row-major, lda; TRANS_A reads A[k*lda+m], i.e. A is [K,M])
//   W: [K,N] f32 (row-major, ldw; strideW==0 -> shared weights)
// Block = 128 threads (4 waves); block tile 64x64; wave tile 16x64; K step 32.
// LDS layouts chosen so every WMMA fragment is contiguous (ds_load_b128):
//   As [m][k] (rows 80B, 16B aligned), Ws2 [n][k] (W tile stored transposed).
// ---------------------------------------------------------------------------
template <bool TRANS_A, bool RELU, bool ACCUM>
__global__ __launch_bounds__(128)
void k_gemm(const float* __restrict__ A, int lda, long strideA,
            const float* __restrict__ W, int ldw, long strideW,
            const float* __restrict__ bias,
            float* __restrict__ out, int ldo, long strideO,
            int M, int N, int K)
{
  __shared__ __bf16 As [64][40];   // [m][k]
  __shared__ __bf16 Ws2[64][40];   // [n][k]  (transposed W tile)

  const int b = blockIdx.z;
  A   += (long)b * strideA;
  W   += (long)b * strideW;
  out += (long)b * strideO;

  const int m0   = blockIdx.y * 64;
  const int n0   = blockIdx.x * 64;
  const int tid  = threadIdx.x;
  const int wave = tid >> 5;
  const int lane = tid & 31;
  const int half = lane >> 4;
  const int colL = lane & 15;

  v8f zero = {};
  v8f acc[4];
#pragma unroll
  for (int t = 0; t < 4; ++t) acc[t] = zero;

  for (int k0 = 0; k0 < K; k0 += 32) {
    // ---- A tile: 64 x 32 (float4 global loads, packed bf16 LDS stores) ----
    if (TRANS_A) {
      // A is [K,M]; contiguous along m.
#pragma unroll
      for (int i = 0; i < 4; ++i) {
        int idx = i * 128 + tid;            // 512 float4
        int c  = idx >> 4;                  // k   0..31
        int r4 = idx & 15;                  // m/4 0..15
        float4 v = *(const float4*)(A + (long)(k0 + c) * lda + (m0 + r4 * 4));
        As[r4 * 4 + 0][c] = (__bf16)v.x;
        As[r4 * 4 + 1][c] = (__bf16)v.y;
        As[r4 * 4 + 2][c] = (__bf16)v.z;
        As[r4 * 4 + 3][c] = (__bf16)v.w;
      }
    } else {
#pragma unroll
      for (int i = 0; i < 4; ++i) {
        int idx = i * 128 + tid;            // 512 float4
        int r  = idx >> 3;                  // m   0..63
        int c4 = idx & 7;                   // k/4 0..7
        float4 v = *(const float4*)(A + (long)(m0 + r) * lda + (k0 + c4 * 4));
        uint2 p; p.x = pk2bf(v.x, v.y); p.y = pk2bf(v.z, v.w);
        *(uint2*)&As[r][c4 * 4] = p;
      }
    }
    // ---- W tile: 32 x 64, stored transposed into Ws2[n][k] ----
#pragma unroll
    for (int i = 0; i < 4; ++i) {
      int idx = i * 128 + tid;              // 512 float4
      int r  = idx >> 4;                    // k   0..31
      int c4 = idx & 15;                    // n/4 0..15
      float4 v = *(const float4*)(W + (long)(k0 + r) * ldw + (n0 + c4 * 4));
      Ws2[c4 * 4 + 0][r] = (__bf16)v.x;
      Ws2[c4 * 4 + 1][r] = (__bf16)v.y;
      Ws2[c4 * 4 + 2][r] = (__bf16)v.z;
      Ws2[c4 * 4 + 3][r] = (__bf16)v.w;
    }
    // prefetch next K-tile while this one is consumed
    if (k0 + 32 < K) {
      int t32 = tid & 31;
      __builtin_prefetch(&W[(long)(k0 + 32 + t32) * ldw + n0 + ((tid >> 5) << 4)], 0, 1);
      if (TRANS_A)
        __builtin_prefetch(&A[(long)(k0 + 32 + t32) * lda + m0 + ((tid >> 5) << 4)], 0, 1);
      else
        __builtin_prefetch(&A[(long)(m0 + (tid >> 1)) * lda + k0 + 32], 0, 1);
    }
    __syncthreads();

    // ---- fragments + WMMA (all contiguous -> ds_load_b128) ----
    v16bf af;
    {
      int row = wave * 16 + colL;
#pragma unroll
      for (int j = 0; j < 16; ++j) {
        int k = (half << 4) + j;
        af[j] = As[row][k];
      }
    }
#pragma unroll
    for (int t = 0; t < 4; ++t) {
      v16bf bf;
      int col = t * 16 + colL;
#pragma unroll
      for (int j = 0; j < 16; ++j)
        bf[j] = Ws2[col][(half << 4) + j];
      acc[t] = wmma_bf16(af, bf, acc[t]);
    }
    __syncthreads();
  }

  // ---- epilogue: bias / accumulate / relu ----
#pragma unroll
  for (int t = 0; t < 4; ++t) {
    int col = n0 + t * 16 + colL;
    float bv = bias ? bias[col] : 0.f;
#pragma unroll
    for (int i = 0; i < 8; ++i) {
      int row = m0 + wave * 16 + half * 8 + i;
      long idx = (long)row * ldo + col;
      float v = acc[t][i] + bv;
      if (ACCUM) v += out[idx];
      if (RELU)  v = v > 0.f ? v : 0.f;
      out[idx] = v;
    }
  }
}

// ---------------------------------------------------------------------------
// Flash attention, H heads of dim 32 packed in QKV[b, n, h*96 + {0,32,64} + c]
// Block = 4 waves; wave owns 16 query rows; keys streamed 32 at a time.
// Ks kept [key][chan] (K^T fragment is contiguous); Vs stored transposed
// [chan][key] so the V fragment is contiguous; P tile rows padded to 40
// halves so the P fragment is 16B-aligned contiguous.
// ---------------------------------------------------------------------------
__global__ __launch_bounds__(128)
void k_attn(const float* __restrict__ QKV, float* __restrict__ OUT, int Ntok)
{
  __shared__ __bf16 Ks [32][40];       // [key][chan]
  __shared__ __bf16 Vs2[32][40];       // [chan][key]
  __shared__ __bf16 Ps[4][16][40];     // per-wave P tile [row][key]

  const int b    = blockIdx.z;
  const int h    = blockIdx.y;
  const int n0   = blockIdx.x * 64;
  const int tid  = threadIdx.x;
  const int wave = tid >> 5;
  const int lane = tid & 31;
  const int half = lane >> 4;
  const int colL = lane & 15;
  const float scale = 0.17677669529663687f;   // 1/sqrt(32)

  const float* base = QKV + (long)b * Ntok * 768 + h * 96;

  // q fragment (rows n0+wave*16 .. +15), pre-scaled
  v16bf qf;
  {
    int row = n0 + wave * 16 + colL;
#pragma unroll
    for (int j = 0; j < 16; ++j) {
      int k = (half << 4) + j;
      qf[j] = (__bf16)(base[(long)row * 768 + k] * scale);
    }
  }

  float m_i[8], l_i[8];
  v8f O0 = {}, O1 = {};
#pragma unroll
  for (int i = 0; i < 8; ++i) { m_i[i] = -1e30f; l_i[i] = 0.f; }

  for (int kc = 0; kc < Ntok; kc += 32) {
    // stage 32 keys of K and V (float4 loads; V stored transposed)
#pragma unroll
    for (int i = 0; i < 2; ++i) {
      int idx = i * 128 + tid;              // 256 float4
      int r  = idx >> 3;                    // key    0..31
      int c4 = idx & 7;                     // chan/4 0..7
      float4 kv = *(const float4*)(base + (long)(kc + r) * 768 + 32 + c4 * 4);
      uint2 p; p.x = pk2bf(kv.x, kv.y); p.y = pk2bf(kv.z, kv.w);
      *(uint2*)&Ks[r][c4 * 4] = p;
      float4 vv = *(const float4*)(base + (long)(kc + r) * 768 + 64 + c4 * 4);
      Vs2[c4 * 4 + 0][r] = (__bf16)vv.x;
      Vs2[c4 * 4 + 1][r] = (__bf16)vv.y;
      Vs2[c4 * 4 + 2][r] = (__bf16)vv.z;
      Vs2[c4 * 4 + 3][r] = (__bf16)vv.w;
    }
    __syncthreads();

    // S = q @ K^T  (B[k=chan][n=key] = Ks[key][chan], contiguous per lane)
    v16bf b0, b1;
#pragma unroll
    for (int j = 0; j < 16; ++j) {
      int k = (half << 4) + j;
      b0[j] = Ks[colL][k];
      b1[j] = Ks[colL + 16][k];
    }
    v8f z = {};
    v8f S0 = wmma_bf16(qf, b0, z);
    v8f S1 = wmma_bf16(qf, b1, z);

    // online softmax per row (reductions across the 16 lanes of a half)
#pragma unroll
    for (int i = 0; i < 8; ++i) {
      float mx = fmaxf(S0[i], S1[i]);
#pragma unroll
      for (int d = 1; d < 16; d <<= 1) mx = fmaxf(mx, __shfl_xor(mx, d, 32));
      float mnew = fmaxf(m_i[i], mx);
      float p0 = __expf(S0[i] - mnew);
      float p1 = __expf(S1[i] - mnew);
      float s = p0 + p1;
#pragma unroll
      for (int d = 1; d < 16; d <<= 1) s += __shfl_xor(s, d, 32);
      float alpha = __expf(m_i[i] - mnew);
      l_i[i] = l_i[i] * alpha + s;
      m_i[i] = mnew;
      O0[i] *= alpha; O1[i] *= alpha;
      Ps[wave][half * 8 + i][colL]      = (__bf16)p0;
      Ps[wave][half * 8 + i][colL + 16] = (__bf16)p1;
    }
    __syncthreads();

    // O += P @ V   (P fragment and V fragments are contiguous)
    v16bf pf;
#pragma unroll
    for (int j = 0; j < 16; ++j) {
      int k = (half << 4) + j;
      pf[j] = Ps[wave][colL][k];
    }
    v16bf vb0, vb1;
#pragma unroll
    for (int j = 0; j < 16; ++j) {
      int kk = (half << 4) + j;
      vb0[j] = Vs2[colL][kk];
      vb1[j] = Vs2[colL + 16][kk];
    }
    O0 = wmma_bf16(pf, vb0, O0);
    O1 = wmma_bf16(pf, vb1, O1);
    __syncthreads();
  }

  float* op = OUT + (long)b * Ntok * 256 + h * 32;
#pragma unroll
  for (int i = 0; i < 8; ++i) {
    int row = n0 + wave * 16 + half * 8 + i;
    float inv = 1.f / l_i[i];
    op[(long)row * 256 + colL]      = O0[i] * inv;
    op[(long)row * 256 + colL + 16] = O1[i] * inv;
  }
}

// ---------------------------------------------------------------------------
// out = LayerNorm(X + Y) * g + beta      (one wave per row, D<=256)
// ---------------------------------------------------------------------------
__global__ __launch_bounds__(128)
void k_add_ln(const float* __restrict__ X, const float* __restrict__ Y,
              const float* __restrict__ g, const float* __restrict__ beta,
              float* __restrict__ out, int D, long nrows)
{
  long row = (long)blockIdx.x * 4 + (threadIdx.x >> 5);
  int lane = threadIdx.x & 31;
  if (row >= nrows) return;
  const float* x = X + row * D;
  const float* y = Y + row * D;
  float* o = out + row * D;
  int per = D / 32;
  float vals[8];
  float s = 0.f, ss = 0.f;
  for (int i = 0; i < per; ++i) {
    float v = x[lane + i * 32] + y[lane + i * 32];
    vals[i] = v; s += v; ss += v * v;
  }
#pragma unroll
  for (int d = 1; d < 32; d <<= 1) {
    s  += __shfl_xor(s,  d, 32);
    ss += __shfl_xor(ss, d, 32);
  }
  float mean = s / D;
  float var  = ss / D - mean * mean;
  float r = rsqrtf(var + 1e-5f);
  for (int i = 0; i < per; ++i) {
    int c = lane + i * 32;
    o[c] = (vals[i] - mean) * r * g[c] + beta[c];
  }
}

// div_n[b,n] = sum_e G[b,e,n]
__global__ void k_colsum(const float* __restrict__ G, float* __restrict__ out,
                         int E, int N)
{
  int b = blockIdx.y;
  int n = blockIdx.x * blockDim.x + threadIdx.x;
  if (n >= N) return;
  const float* g = G + (long)b * E * N + n;
  float s = 0.f;
  for (int e = 0; e < E; ++e) s += g[(long)e * N];
  out[b * N + n] = s;
}

// div_e[row] = sum_n G[row, n]   (one wave per row)
__global__ __launch_bounds__(128)
void k_rowsum(const float* __restrict__ G, float* __restrict__ out,
              int N, long nrows)
{
  long row = (long)blockIdx.x * 4 + (threadIdx.x >> 5);
  int lane = threadIdx.x & 31;
  if (row >= nrows) return;
  const float* g = G + row * N;
  float s = 0.f;
  for (int c = lane; c < N; c += 32) s += g[c];
#pragma unroll
  for (int d = 1; d < 32; d <<= 1) s += __shfl_xor(s, d, 32);
  if (lane == 0) out[row] = s;
}

// out[i] = (X ? X[i] : 0) + Y[i] / S[i / C]
__global__ void k_scale_add(const float* __restrict__ X, const float* __restrict__ Y,
                            const float* __restrict__ S, float* __restrict__ out,
                            int C, long total)
{
  long i = (long)blockIdx.x * blockDim.x + threadIdx.x;
  if (i >= total) return;
  long r = i / C;
  float v = Y[i] / S[r];
  if (X) v += X[i];
  out[i] = v;
}

// ---------------------------------------------------------------------------
// Host side
// ---------------------------------------------------------------------------
static void gemm(bool transA, bool relu, bool accum, hipStream_t st, int B,
                 const float* A, int lda, long sA,
                 const float* W, int ldw, long sW, const float* bias,
                 float* out, int ldo, long sO, int M, int Nc, int K)
{
  dim3 g(Nc / 64, M / 64, B), blk(128);
  if (transA)
    k_gemm<true,  false, false><<<g, blk, 0, st>>>(A, lda, sA, W, ldw, sW, bias, out, ldo, sO, M, Nc, K);
  else if (relu && accum)
    k_gemm<false, true,  true ><<<g, blk, 0, st>>>(A, lda, sA, W, ldw, sW, bias, out, ldo, sO, M, Nc, K);
  else if (relu)
    k_gemm<false, true,  false><<<g, blk, 0, st>>>(A, lda, sA, W, ldw, sW, bias, out, ldo, sO, M, Nc, K);
  else
    k_gemm<false, false, false><<<g, blk, 0, st>>>(A, lda, sA, W, ldw, sW, bias, out, ldo, sO, M, Nc, K);
}

extern "C" void kernel_launch(void* const* d_in, const int* in_sizes, int n_in,
                              void* d_out, int out_size, void* d_ws, size_t ws_size,
                              hipStream_t stream)
{
  (void)in_sizes; (void)n_in; (void)out_size; (void)ws_size;
  const int B = 8, N = 1024, E = 4096, ND = 256, ED = 256, H = 8;
  const int NH = 1024, EH = 1024;

  const float* nf_in = (const float*)d_in[0];
  const float* ef_in = (const float*)d_in[1];
  const float* G     = (const float*)d_in[2];

  float* ws = (float*)d_ws;
  size_t off = 0;
  auto take = [&](size_t n) { float* p = ws + off; off += n; return p; };
  float* r_nf   = take((size_t)B * N * ND);        // node features
  float* r_ef   = take((size_t)B * E * ED);        // edge features
  float* r_qkvn = take((size_t)B * N * 768);       // node QKV / combined QKV
  float* r_big  = take((size_t)B * E * 1024);      // edge QKV (768) then edge hidden (1024)
  float* r_agg  = take((size_t)B * N * 768);       // G^T @ qkv_e
  float* r_att  = take((size_t)B * N * ND);        // attention output
  float* r_tmpn = take((size_t)B * N * ND);        // node temp
  float* r_sh   = take((size_t)B * E * ND);        // hid_n (B*N*NH) / M=G@nf / edge temp
  float* r_divn = take((size_t)B * N);
  float* r_dive = take((size_t)B * E);

  hipMemcpyAsync(r_nf, nf_in, (size_t)B * N * ND * 4, hipMemcpyDeviceToDevice, stream);
  hipMemcpyAsync(r_ef, ef_in, (size_t)B * E * ED * 4, hipMemcpyDeviceToDevice, stream);

  // G is constant across layers: denominators once
  k_colsum<<<dim3(N / 256, B), 256, 0, stream>>>(G, r_divn, E, N);
  k_rowsum<<<dim3((B * E) / 4), 128, 0, stream>>>(G, r_dive, N, (long)B * E);

  for (int layer = 0; layer < 3; ++layer) {
    // pytree leaves per layer, dict keys sorted:
    const float* const* P = (const float* const*)(d_in + 3 + layer * 26);
    const float* We1a = P[0];  const float* We1b = P[1];
    const float* We2a = P[2];  const float* We2b = P[3];
    const float* Wn1  = P[4];  const float* Wn2  = P[5];
    const float* Wo   = P[6];  const float* Wqe  = P[7];  const float* Wqn = P[8];
    const float* b1e  = P[9];  const float* b1n  = P[10];
    const float* b2e  = P[11]; const float* b2n  = P[12];
    const float* be1a = P[13]; const float* be1b = P[14];
    const float* be2a = P[15]; const float* be2b = P[16];
    const float* bn1  = P[17]; const float* bn2  = P[18]; const float* bo = P[19];
    const float* bqe  = P[20]; const float* bqn  = P[21];
    const float* g1e  = P[22]; const float* g1n  = P[23];
    const float* g2e  = P[24]; const float* g2n  = P[25];

    // ---- attention with edge-aggregated QKV ----
    gemm(false, false, false, stream, B, r_nf, ND, (long)N * ND, Wqn, 768, 0, bqn,
         r_qkvn, 768, (long)N * 768, N, 768, ND);
    gemm(false, false, false, stream, B, r_ef, ED, (long)E * ED, Wqe, 768, 0, bqe,
         r_big, 768, (long)E * 768, E, 768, ED);
    // AGG = G^T @ qkv_e   (A = G read transposed, B = per-batch qkv_e)
    gemm(true, false, false, stream, B, G, N, (long)E * N, r_big, 768, (long)E * 768,
         nullptr, r_agg, 768, (long)N * 768, N, 768, E);
    // QKV = qkv_n + AGG / div_n
    {
      long total = (long)B * N * 768;
      k_scale_add<<<dim3((total + 255) / 256), 256, 0, stream>>>(r_qkvn, r_agg, r_divn,
                                                                 r_qkvn, 768, total);
    }
    k_attn<<<dim3(N / 64, H, B), 128, 0, stream>>>(r_qkvn, r_att, N);
    gemm(false, false, false, stream, B, r_att, ND, (long)N * ND, Wo, ND, 0, bo,
         r_tmpn, ND, (long)N * ND, N, ND, ND);
    k_add_ln<<<dim3((B * N) / 4), 128, 0, stream>>>(r_nf, r_tmpn, g1n, b1n, r_nf, ND, (long)B * N);

    // ---- node FFN ----
    gemm(false, true, false, stream, B, r_nf, ND, (long)N * ND, Wn1, NH, 0, bn1,
         r_sh, NH, (long)N * NH, N, NH, ND);
    gemm(false, false, false, stream, B, r_sh, NH, (long)N * NH, Wn2, ND, 0, bn2,
         r_tmpn, ND, (long)N * ND, N, ND, NH);
    k_add_ln<<<dim3((B * N) / 4), 128, 0, stream>>>(r_nf, r_tmpn, g2n, b2n, r_nf, ND, (long)B * N);

    // ---- edge update ----
    // M = (G @ nf) / div_e   (batched B = nf)
    gemm(false, false, false, stream, B, G, N, (long)E * N, r_nf, ND, (long)N * ND,
         nullptr, r_sh, ND, (long)E * ND, E, ND, N);
    {
      long total = (long)B * E * ND;
      k_scale_add<<<dim3((total + 255) / 256), 256, 0, stream>>>(nullptr, r_sh, r_dive,
                                                                 r_sh, ND, total);
    }
    // hid = relu([ef, M] @ We1a + be1a)  as two GEMMs (second accumulates + relu)
    gemm(false, false, false, stream, B, r_ef, ED, (long)E * ED, We1a, EH, 0, be1a,
         r_big, EH, (long)E * EH, E, EH, ED);
    gemm(false, true, true, stream, B, r_sh, ND, (long)E * ND, We1a + (size_t)ED * EH, EH, 0,
         nullptr, r_big, EH, (long)E * EH, E, EH, ND);
    gemm(false, false, false, stream, B, r_big, EH, (long)E * EH, We1b, ED, 0, be1b,
         r_sh, ED, (long)E * ED, E, ED, EH);
    k_add_ln<<<dim3((B * E) / 4), 128, 0, stream>>>(r_ef, r_sh, g1e, b1e, r_ef, ED, (long)B * E);

    gemm(false, true, false, stream, B, r_ef, ED, (long)E * ED, We2a, EH, 0, be2a,
         r_big, EH, (long)E * EH, E, EH, ED);
    gemm(false, false, false, stream, B, r_big, EH, (long)E * EH, We2b, ED, 0, be2b,
         r_sh, ED, (long)E * ED, E, ED, EH);
    k_add_ln<<<dim3((B * E) / 4), 128, 0, stream>>>(r_ef, r_sh, g2e, b2e, r_ef, ED, (long)B * E);
  }

  // output tuple (nf, G)
  float* out = (float*)d_out;
  hipMemcpyAsync(out, r_nf, (size_t)B * N * ND * 4, hipMemcpyDeviceToDevice, stream);
  hipMemcpyAsync(out + (size_t)B * N * ND, G, (size_t)B * E * N * 4,
                 hipMemcpyDeviceToDevice, stream);
}
